// DualMPNN_8985071583716
// MI455X (gfx1250) — matrix operand order repaired and checked by reading, not compiled, not run on hardware
//
#include <hip/hip_runtime.h>
#include <hip/hip_bf16.h>

// ---------------- problem constants (match reference) ----------------
#define kN       100000
#define kE       800000
#define kG       512
#define kH       64
#define kMSG     128
#define kNODEIN  92
#define kEDGEIN  16
#define kL       3

// ---------------- WMMA types ----------------
typedef __attribute__((ext_vector_type(16))) __bf16 v16bf;
typedef __attribute__((ext_vector_type(8)))  float  v8f;

union ABf { v16bf v; unsigned u[8]; };

// exact RNE (cold paths: weight repack)
static __device__ __forceinline__ unsigned short f2bf(float x) {
    union { float f; unsigned u; } c; c.f = x;
    unsigned r = c.u + 0x7FFFu + ((c.u >> 16) & 1u);
    return (unsigned short)(r >> 16);
}
static __device__ __forceinline__ unsigned packbf(float lo, float hi) {
    return (unsigned)f2bf(lo) | ((unsigned)f2bf(hi) << 16);
}
// fast round-half-up + v_perm pack (hot paths)
static __device__ __forceinline__ unsigned packbf_fast(float lo, float hi) {
    union { float f; unsigned u; } a, b; a.f = lo; b.f = hi;
    return __builtin_amdgcn_perm(b.u + 0x8000u, a.u + 0x8000u, 0x07060302u);
}
static __device__ __forceinline__ unsigned short f2bf_fast(float x) {
    union { float f; unsigned u; } c; c.f = x;
    return (unsigned short)((c.u + 0x8000u) >> 16);
}

// --------------------------------------------------------------------
// Fragment-major storage: a [16 x K] bf16 A-tile (or [K x 16] B-tile) is
// stored as, per 32-K chunk, 256 dwords: lane L owns dwords [L*8 .. L*8+7]
// (its 8 WMMA source VGPRs, 32 B contiguous -> b128 loads).
//
// A layout (ISA 7.12.2): element (m, k):  lane = m + 16*((k>>3)&1),
//   vgpr = 4*((k>>4)&1) + ((k>>1)&3),  bit0 of k selects halfword.
// --------------------------------------------------------------------
static __device__ __forceinline__ int a_idx(int m, int k) {   // dword index (k even pair)
    return ((k >> 5) << 8) + ((m + (((k >> 3) & 1) << 4)) << 3)
         + (((k >> 4) & 1) << 2) + ((k >> 1) & 3);
}

static __device__ __forceinline__ v16bf load_frag_fm(const unsigned* base, int fragIdx, int lane) {
    const uint4* p = (const uint4*)(base + (fragIdx << 8) + (lane << 3));
    uint4 a = p[0], b = p[1];
    ABf f;
    f.u[0] = a.x; f.u[1] = a.y; f.u[2] = a.z; f.u[3] = a.w;
    f.u[4] = b.x; f.u[5] = b.y; f.u[6] = b.z; f.u[7] = b.w;
    return f.v;
}

#define WMMA_BF16(a, b, c) \
    __builtin_amdgcn_wmma_f32_16x16x32_bf16(false, (a), false, (b), (short)0, (c), false, false)

// ---------------- weight repack: fp32 [K][Nn] -> bf16 fragment-major ----------------
// B layout (ISA 7.12.4): lane L -> N = t*16 + (L&15); vgpr v -> K = c*32 + 8*(L>>4) + 16*(v>>2) + 2*(v&3).
__global__ void pack_w_fm_kernel(const float* __restrict__ src, unsigned* __restrict__ dst,
                                 int K, int Nn) {
    int idx = blockIdx.x * blockDim.x + threadIdx.x;   // one output dword
    int nT = Nn >> 4;
    int total = (K >> 5) * nT * 256;
    if (idx >= total) return;
    int f = idx >> 8, rem = idx & 255;
    int L = rem >> 3, v = rem & 7;
    int c = f / nT, t = f - c * nT;
    int n = t * 16 + (L & 15);
    int k = c * 32 + ((L >> 4) << 3) + ((v >> 2) << 4) + ((v & 3) << 1);
    dst[idx] = packbf(src[k * Nn + n], src[(k + 1) * Nn + n]);
}

// ---------------- embeddings ----------------
__global__ void node_embed_kernel(const float* __restrict__ x, const float* __restrict__ w,
                                  const float* __restrict__ b, float* __restrict__ h) {
    __shared__ float sx[kNODEIN];
    int n = blockIdx.x;              // one node per 64-thread block
    int j = threadIdx.x;
    for (int k = j; k < kNODEIN; k += 64) sx[k] = x[(size_t)n * kNODEIN + k];
    __syncthreads();
    float acc = b[j];
#pragma unroll 4
    for (int k = 0; k < kNODEIN; ++k) acc = fmaf(sx[k], w[k * kH + j], acc);
    h[(size_t)n * kH + j] = acc > 0.f ? acc : 0.f;
}

__global__ void edge_embed_kernel(const float* __restrict__ ea, const float* __restrict__ w,
                                  const float* __restrict__ b, unsigned short* __restrict__ eaB) {
    __shared__ float se[4][kEDGEIN];
    int le = threadIdx.x >> 6, j = threadIdx.x & 63;
    int e = blockIdx.x * 4 + le;
    if (j < kEDGEIN && e < kE) se[le][j] = ea[(size_t)e * kEDGEIN + j];
    __syncthreads();
    if (e >= kE) return;
    float acc = b[j];
#pragma unroll
    for (int k = 0; k < kEDGEIN; ++k) acc = fmaf(se[le][k], w[k * kH + j], acc);
    eaB[(size_t)e * kH + j] = f2bf_fast(acc > 0.f ? acc : 0.f);
}

// ---------------- edge MLP + scatter (the WMMA workhorse) ----------------
__global__ __launch_bounds__(256)
void edge_msg_kernel(const float* __restrict__ h,
                     const unsigned short* __restrict__ eaB,
                     const long long* __restrict__ src,
                     const long long* __restrict__ dst,
                     const unsigned* __restrict__ w1p, const float* __restrict__ b1,
                     const unsigned* __restrict__ w2p, const float* __restrict__ b2,
                     float* __restrict__ agg) {
    // per wave: 16x128 bf16 A-tile, fragment-major: 4 chunks x 256 dwords = 4 KB
    __shared__ __align__(16) unsigned sT[8][1024];
    int wv = threadIdx.x >> 5, lane = threadIdx.x & 31;
    int eBase = blockIdx.x * 128 + wv * 16;           // E % 128 == 0 -> always in range
    unsigned* tile = sT[wv];

    // Stage A tile: cols 0..63 = h[src[e]] (lanes 0-15, one row each),
    //               cols 64..127 = ea[e]   (lanes 16-31, one row each)
    if (lane < 16) {
        int m = lane;
        long long s = src[eBase + m];
        const float* hr = h + (size_t)s * kH;
#pragma unroll
        for (int c = 0; c < kH; c += 4) {
            float4 v = *(const float4*)(hr + c);
            tile[a_idx(m, c)]     = packbf_fast(v.x, v.y);
            tile[a_idx(m, c + 2)] = packbf_fast(v.z, v.w);
        }
    } else {
        int m = lane & 15;
        const uint4* er = (const uint4*)(eaB + (size_t)(eBase + m) * kH);
#pragma unroll
        for (int q = 0; q < 8; ++q) {                 // 8 x uint4 = 32 bf16-pairs
            uint4 v = er[q];
            int k = kH + (q << 3);                    // global col of first pair
            tile[a_idx(m, k)]     = v.x;
            tile[a_idx(m, k + 2)] = v.y;
            tile[a_idx(m, k + 4)] = v.z;
            tile[a_idx(m, k + 6)] = v.w;
        }
    }
    __syncthreads();

    // GEMM1: [16x128] x [128x128] -> 8 accumulator tiles
    v8f acc[8] = {};
#pragma unroll
    for (int c = 0; c < 4; ++c) {
        v16bf a = load_frag_fm(tile, c, lane);
#pragma unroll
        for (int t = 0; t < 8; ++t) {
            v16bf b = load_frag_fm(w1p, c * 8 + t, lane);
            acc[t] = WMMA_BF16(a, b, acc[t]);
        }
    }
    __syncthreads();                                  // before overwriting the LDS tile

    // bias + ReLU, write hidden back as bf16 A-tile (halfword stores: adjacent K
    // of the next GEMM live in adjacent lanes here, so pairs can't form locally)
    {
        unsigned short* tile16 = (unsigned short*)tile;
        int nc = lane & 15;
        int hf8 = (lane >> 4) << 3;                   // C/D layout: vgpr i -> M = i + 8*(lane>=16)
#pragma unroll
        for (int t = 0; t < 8; ++t) {
            float bb = b1[t * 16 + nc];
            int kcol = t * 16 + nc;
#pragma unroll
            for (int i = 0; i < 8; ++i) {
                float v = acc[t][i] + bb;
                v = v > 0.f ? v : 0.f;
                tile16[(a_idx(i + hf8, kcol) << 1) + (kcol & 1)] = f2bf_fast(v);
            }
        }
    }
    __syncthreads();

    // GEMM2: [16x128] x [128x64]
    v8f acc2[4] = {};
#pragma unroll
    for (int c = 0; c < 4; ++c) {
        v16bf a = load_frag_fm(tile, c, lane);
#pragma unroll
        for (int t = 0; t < 4; ++t) {
            v16bf b = load_frag_fm(w2p, c * 4 + t, lane);
            acc2[t] = WMMA_BF16(a, b, acc2[t]);
        }
    }

    // bias + atomic scatter-add into agg[dst]
    {
        int nc = lane & 15;
        int hf8 = (lane >> 4) << 3;
        long long dd[8];
#pragma unroll
        for (int i = 0; i < 8; ++i) dd[i] = dst[eBase + i + hf8];
#pragma unroll
        for (int t = 0; t < 4; ++t) {
            float bb = b2[t * 16 + nc];
#pragma unroll
            for (int i = 0; i < 8; ++i)
                atomicAdd(agg + (size_t)dd[i] * kH + t * 16 + nc, acc2[t][i] + bb);
        }
    }
}

// ---------------- node update: h' = relu(agg + h@self_w + b) ----------------
__global__ __launch_bounds__(256)
void node_update_kernel(const float* __restrict__ hin, const float* __restrict__ agg,
                        const unsigned* __restrict__ wsp, const float* __restrict__ sb,
                        float* __restrict__ hout) {
    __shared__ __align__(16) unsigned sT[8][512];     // 16x64 bf16, 2 chunks x 256 dwords
    int wv = threadIdx.x >> 5, lane = threadIdx.x & 31;
    int nBase = blockIdx.x * 128 + wv * 16;
    unsigned* tile = sT[wv];
    {
        int m = lane & 15, half = lane >> 4;          // each lane stages half a row (32 cols)
        int node = nBase + m;
        int kb = half * 32;
        if (node < kN) {
            const float* hr = hin + (size_t)node * kH + kb;
#pragma unroll
            for (int c = 0; c < 32; c += 4) {
                float4 v = *(const float4*)(hr + c);
                tile[a_idx(m, kb + c)]     = packbf_fast(v.x, v.y);
                tile[a_idx(m, kb + c + 2)] = packbf_fast(v.z, v.w);
            }
        } else {
#pragma unroll
            for (int c = 0; c < 32; c += 2) tile[a_idx(m, kb + c)] = 0;
        }
    }
    __syncthreads();

    v8f acc[4] = {};
#pragma unroll
    for (int c = 0; c < 2; ++c) {
        v16bf a = load_frag_fm(tile, c, lane);
#pragma unroll
        for (int t = 0; t < 4; ++t) {
            v16bf b = load_frag_fm(wsp, c * 4 + t, lane);
            acc[t] = WMMA_BF16(a, b, acc[t]);
        }
    }
    int nc = lane & 15, hf8 = (lane >> 4) << 3;
#pragma unroll
    for (int t = 0; t < 4; ++t) {
        float bb = sb[t * 16 + nc];
#pragma unroll
        for (int i = 0; i < 8; ++i) {
            int node = nBase + i + hf8;
            if (node < kN) {
                int col = t * 16 + nc;
                float v = acc[t][i] + bb + agg[(size_t)node * kH + col];
                hout[(size_t)node * kH + col] = v > 0.f ? v : 0.f;
            }
        }
    }
}

// ---------------- pooling + readout ----------------
__global__ void pool_kernel(const float* __restrict__ h, const long long* __restrict__ batch,
                            float* __restrict__ gsum, float* __restrict__ cnt) {
    int idx = blockIdx.x * blockDim.x + threadIdx.x;
    int n = idx >> 6, j = idx & 63;
    if (n >= kN) return;
    long long g = batch[n];
    atomicAdd(gsum + (size_t)g * kH + j, h[(size_t)n * kH + j]);
    if (j == 0) atomicAdd(cnt + g, 1.0f);
}

__global__ void readout_kernel(const float* __restrict__ gsum, const float* __restrict__ cnt,
                               const float* __restrict__ w1, const float* __restrict__ b1,
                               const float* __restrict__ w2, const float* __restrict__ b2,
                               float* __restrict__ out) {
    __shared__ float sg[2 * kH];
    __shared__ float sh[kH];
    int g = blockIdx.x, j = threadIdx.x;     // 64 threads
    float c = cnt[g]; c = c < 1.f ? 1.f : c;
    float v = gsum[(size_t)g * kH + j];
    sg[j] = v; sg[kH + j] = v / c;
    __syncthreads();
    float acc = b1[j];
#pragma unroll 4
    for (int k = 0; k < 2 * kH; ++k) acc = fmaf(sg[k], w1[k * kH + j], acc);
    sh[j] = (acc > 0.f ? acc : 0.f) * w2[j];
    __syncthreads();
    if (j == 0) {
        float s = b2[0];
        for (int k = 0; k < kH; ++k) s += sh[k];
        out[g] = s;
    }
}

// ---------------- launcher ----------------
extern "C" void kernel_launch(void* const* d_in, const int* in_sizes, int n_in,
                              void* d_out, int out_size, void* d_ws, size_t ws_size,
                              hipStream_t stream) {
    const float*     x      = (const float*)d_in[0];
    const long long* ei     = (const long long*)d_in[1];
    const float*     eattr  = (const float*)d_in[2];
    const long long* batch  = (const long long*)d_in[3];
    const float*     node_w = (const float*)d_in[4];
    const float*     node_b = (const float*)d_in[5];
    const float*     edge_w = (const float*)d_in[6];
    const float*     edge_b = (const float*)d_in[7];
    const float*     msg_w1 = (const float*)d_in[8];
    const float*     msg_b1 = (const float*)d_in[9];
    const float*     msg_w2 = (const float*)d_in[10];
    const float*     msg_b2 = (const float*)d_in[11];
    const float*     self_w = (const float*)d_in[12];
    const float*     self_b = (const float*)d_in[13];
    const float*     ro_w1  = (const float*)d_in[14];
    const float*     ro_b1  = (const float*)d_in[15];
    const float*     ro_w2  = (const float*)d_in[16];
    const float*     ro_b2  = (const float*)d_in[17];
    float* out = (float*)d_out;

    char* ws = (char*)d_ws;
    size_t off = 0;
    auto alloc = [&](size_t bytes) -> void* {
        void* p = ws + off;
        off = (off + bytes + 255) & ~(size_t)255;
        return p;
    };
    float*    hA   = (float*)alloc(sizeof(float) * (size_t)kN * kH);
    float*    hB   = (float*)alloc(sizeof(float) * (size_t)kN * kH);
    float*    agg  = (float*)alloc(sizeof(float) * (size_t)kN * kH);
    unsigned short* eaB = (unsigned short*)alloc(sizeof(unsigned short) * (size_t)kE * kH);
    float*    gsum = (float*)alloc(sizeof(float) * (size_t)kG * kH);
    float*    cnt  = (float*)alloc(sizeof(float) * kG);
    // fragment-major packed weights (dwords): w1 32 frags, w2 16 frags, self 8 frags / layer
    unsigned* w1p  = (unsigned*)alloc(sizeof(unsigned) * kL * 32 * 256);
    unsigned* w2p  = (unsigned*)alloc(sizeof(unsigned) * kL * 16 * 256);
    unsigned* swp  = (unsigned*)alloc(sizeof(unsigned) * kL * 8 * 256);

    for (int l = 0; l < kL; ++l) {
        pack_w_fm_kernel<<<(32 * 256 + 255) / 256, 256, 0, stream>>>(
            msg_w1 + (size_t)l * 2 * kH * kMSG, w1p + (size_t)l * 32 * 256, 2 * kH, kMSG);
        pack_w_fm_kernel<<<(16 * 256 + 255) / 256, 256, 0, stream>>>(
            msg_w2 + (size_t)l * kMSG * kH, w2p + (size_t)l * 16 * 256, kMSG, kH);
        pack_w_fm_kernel<<<(8 * 256 + 255) / 256, 256, 0, stream>>>(
            self_w + (size_t)l * kH * kH, swp + (size_t)l * 8 * 256, kH, kH);
    }

    node_embed_kernel<<<kN, 64, 0, stream>>>(x, node_w, node_b, hA);
    edge_embed_kernel<<<(kE + 3) / 4, 256, 0, stream>>>(eattr, edge_w, edge_b, eaB);

    const long long* srcI = ei;
    const long long* dstI = ei + kE;
    float* hc = hA;
    float* hn = hB;
    for (int l = 0; l < kL; ++l) {
        hipMemsetAsync(agg, 0, sizeof(float) * (size_t)kN * kH, stream);
        edge_msg_kernel<<<kE / 128, 256, 0, stream>>>(
            hc, eaB, srcI, dstI,
            w1p + (size_t)l * 32 * 256, msg_b1 + l * kMSG,
            w2p + (size_t)l * 16 * 256, msg_b2 + l * kH,
            agg);
        node_update_kernel<<<(kN + 127) / 128, 256, 0, stream>>>(
            hc, agg, swp + (size_t)l * 8 * 256, self_b + l * kH, hn);
        float* t = hc; hc = hn; hn = t;
    }

    hipMemsetAsync(gsum, 0, sizeof(float) * (size_t)kG * kH, stream);
    hipMemsetAsync(cnt, 0, sizeof(float) * kG, stream);
    pool_kernel<<<((size_t)kN * kH + 255) / 256, 256, 0, stream>>>(hc, batch, gsum, cnt);
    readout_kernel<<<kG, 64, 0, stream>>>(gsum, cnt, ro_w1, ro_b1, ro_w2, ro_b2, out);
}
